// AttentionEncoder_77481210020168
// MI455X (gfx1250) — matrix operand order repaired and checked by reading, not compiled, not run on hardware
//
#include <hip/hip_runtime.h>
#include <hip/hip_bf16.h>
#include <math.h>

// Problem constants (from reference): B=32, S=2048, H=1024
#define BB 32
#define SS 2048
#define HH 1024

typedef __attribute__((ext_vector_type(16))) __bf16 v16bf;
typedef __attribute__((ext_vector_type(8)))  __bf16 v8bf;
typedef __attribute__((ext_vector_type(4)))  __bf16 v4bf;
typedef __attribute__((ext_vector_type(8)))  float  v8f;

// ---------------------------------------------------------------------------
// Kernel 0: zero the per-batch L1-sum accumulators
// ---------------------------------------------------------------------------
__global__ void init_sums_kernel(float* __restrict__ sumb) {
    if (threadIdx.x < BB) sumb[threadIdx.x] = 0.0f;
}

// ---------------------------------------------------------------------------
// Kernel 0b: one-shot convert W_en fp32 -> bf16 (2 MB, stays L2-resident).
// ---------------------------------------------------------------------------
__global__ __launch_bounds__(256) void convert_wen_kernel(
    const float* __restrict__ W_en, __bf16* __restrict__ W_en_bf) {
    const long i = ((long)blockIdx.x * 256 + threadIdx.x) * 4;
    float4 v = *(const float4*)(W_en + i);
    v4bf o;
    o[0] = (__bf16)v.x; o[1] = (__bf16)v.y;
    o[2] = (__bf16)v.z; o[3] = (__bf16)v.w;
    *(v4bf*)(W_en_bf + i) = o;
}

// ---------------------------------------------------------------------------
// Kernel 1: debias[b,k] = dot(dehy[b,:], W_de[k,:]) + b_en[k]   (32x1024)
// ---------------------------------------------------------------------------
__global__ __launch_bounds__(256) void debias_kernel(
    const float* __restrict__ dehy, const float* __restrict__ W_de,
    const float* __restrict__ b_en, float* __restrict__ debias) {
    __shared__ float dsh[HH];
    const int b  = blockIdx.x >> 2;
    const int kc = blockIdx.x & 3;
    const int k  = kc * 256 + threadIdx.x;

    for (int i = threadIdx.x; i < HH; i += 256) dsh[i] = dehy[b * HH + i];
    __syncthreads();

    float acc = b_en[k];
    const float* wr = &W_de[(long)k * HH];
    for (int h = 0; h < HH; h += 4) {
        float4 w = *(const float4*)(wr + h);
        acc += dsh[h + 0] * w.x + dsh[h + 1] * w.y +
               dsh[h + 2] * w.z + dsh[h + 3] * w.w;
    }
    debias[b * HH + k] = acc;
}

// ---------------------------------------------------------------------------
// Kernel 2: fused WMMA GEMM + tanh + warp projection + exp + coverage
//   One block per 16-row M-tile of the flattened (B*S, H) problem.
//   8 waves; wave w owns N columns [w*128, w*128+128) = 8 N-tiles of 16.
//   K loop over H in steps of 64: stage 16x64 A tile (one float4 per thread,
//   fully vectorized b128 load -> b64 LDS store), then 2 x 8 WMMAs.
// ---------------------------------------------------------------------------
__global__ __launch_bounds__(256) void fused_score_kernel(
    const float*  __restrict__ enhy,      // [B*S, H] fp32
    const __bf16* __restrict__ W_en_bf,   // [H, H]   bf16 (row n = out col n)
    const float*  __restrict__ W_warp,    // [H]
    const float*  __restrict__ past_attn, // [B*S]
    const float*  __restrict__ debias,    // [B, H]
    float* __restrict__ attn_ee_out,      // [B*S]
    float* __restrict__ attn_un,          // [B*S]  (workspace)
    float* __restrict__ sumb)             // [B]    (global atomics)
{
    // 16x64 A tile as bf16; row stride 72 elems = 144 B (16-B aligned)
    __shared__ __bf16 a_lds[16][72];
    __shared__ float  score_lds[16];

    const int  tid   = threadIdx.x;
    const int  wave  = tid >> 5;
    const int  lane  = tid & 31;
    const int  lrow  = lane & 15;  // M (for A) / N (for B) within 16x16 tile
    const int  lhalf = lane >> 4;  // selects K sub-chunk
    const long mbase = (long)blockIdx.x * 16;   // flattened row base
    const int  b     = (int)(mbase / SS);       // 16 | S, so one batch per block

    // Staging coords: one float4 per thread covers the whole 16x64 tile
    const int  sr = tid >> 4;            // row 0..15
    const int  sc = (tid & 15) * 4;      // col 0..60
    const float* a_src = &enhy[(mbase + sr) * HH + sc];

    v8f acc[8];
#pragma unroll
    for (int i = 0; i < 8; ++i) acc[i] = (v8f){0.f,0.f,0.f,0.f,0.f,0.f,0.f,0.f};

    for (int k0 = 0; k0 < HH; k0 += 64) {
        __syncthreads();
        {
            float4 v = *(const float4*)(a_src + k0);
            v4bf o;
            o[0] = (__bf16)v.x; o[1] = (__bf16)v.y;
            o[2] = (__bf16)v.z; o[3] = (__bf16)v.w;
            *(v4bf*)&a_lds[sr][sc] = o;
            if (k0 + 64 < HH)
                __builtin_prefetch(a_src + k0 + 64, 0, 0);  // global_prefetch_b8
        }
        __syncthreads();

        // Two K=32 WMMA sub-steps off the staged 16x64 tile
#pragma unroll
        for (int kk = 0; kk < 64; kk += 32) {
            // A fragment (contiguous K order used here, same as B below):
            // lane group 0: K = {0..15}; group 1: {16..31}
            v16bf afrag;
            {
                const int kb = kk + lhalf * 16;
#pragma unroll
                for (int j = 0; j < 8; ++j) afrag[j]     = a_lds[lrow][kb + j];
#pragma unroll
                for (int j = 0; j < 8; ++j) afrag[8 + j] = a_lds[lrow][kb + 8 + j];
            }

            // 8 N-tiles per wave; B fragment: 16 contiguous bf16 K-values of
            // W_en row n, K base = 16*lhalf. Two b128 loads per tile.
#pragma unroll
            for (int nt = 0; nt < 8; ++nt) {
                const int n = wave * 128 + nt * 16 + lrow;
                const v8bf* wrow =
                    (const v8bf*)&W_en_bf[(long)n * HH + k0 + kk + 16 * lhalf];
                const v8bf b0 = wrow[0];
                const v8bf b1 = wrow[1];
                v16bf bfrag;
#pragma unroll
                for (int j = 0; j < 8; ++j) { bfrag[j] = b0[j]; bfrag[8 + j] = b1[j]; }

                acc[nt] = __builtin_amdgcn_wmma_f32_16x16x32_bf16(
                    false, afrag, false, bfrag, (short)0, acc[nt], false, false);
            }
        }
    }

    // Epilogue: score[m] = sum_n W_warp[n] * tanh(c[m,n] + debias[b,n])
    // C layout: element j of acc -> M = j + 8*lhalf, N = lrow (+tile offset)
    float partial[8];
#pragma unroll
    for (int j = 0; j < 8; ++j) partial[j] = 0.f;

#pragma unroll
    for (int nt = 0; nt < 8; ++nt) {
        const int n  = wave * 128 + nt * 16 + lrow;
        const float wv = W_warp[n];
        const float db = debias[b * HH + n];
#pragma unroll
        for (int j = 0; j < 8; ++j)
            partial[j] += wv * tanhf(acc[nt][j] + db);
    }

    if (tid < 16) score_lds[tid] = 0.f;
    __syncthreads();
#pragma unroll
    for (int j = 0; j < 8; ++j)
        atomicAdd(&score_lds[j + 8 * lhalf], partial[j]);
    __syncthreads();

    if (tid < 16) {
        const long row = mbase + tid;
        const float e  = expf(score_lds[tid]);
        attn_ee_out[row] = e;
        const float au = e / past_attn[row];
        attn_un[row]   = au;                 // > 0, so |au| == au
        atomicAdd(&sumb[b], au);
    }
}

// ---------------------------------------------------------------------------
// Kernel 3: c_encoder[b,h] = (1/sum[b]) * sum_s attn_un[b,s] * enhy[b,s,h]
// ---------------------------------------------------------------------------
__global__ __launch_bounds__(256) void context_kernel(
    const float* __restrict__ enhy, const float* __restrict__ attn_un,
    const float* __restrict__ sumb, float* __restrict__ c_out) {
    __shared__ float a_sh[256];
    const int b  = blockIdx.x >> 2;
    const int hc = blockIdx.x & 3;
    const int h  = hc * 256 + threadIdx.x;

    float acc = 0.f;
    for (int s0 = 0; s0 < SS; s0 += 256) {
        __syncthreads();
        a_sh[threadIdx.x] = attn_un[b * SS + s0 + threadIdx.x];
        __syncthreads();
        const float* erow = &enhy[((long)b * SS + s0) * HH + h];
        for (int s = 0; s < 256; ++s)
            acc += a_sh[s] * erow[(long)s * HH];
    }
    c_out[b * HH + h] = acc / sumb[b];
}

// ---------------------------------------------------------------------------
// Kernel 4: attn[b,s] = attn_un[b,s] / sum[b]
// ---------------------------------------------------------------------------
__global__ void attn_norm_kernel(const float* __restrict__ attn_un,
                                 const float* __restrict__ sumb,
                                 float* __restrict__ attn_out) {
    const int i = blockIdx.x * 256 + threadIdx.x;
    attn_out[i] = attn_un[i] / sumb[i >> 11];   // S = 2048
}

// ---------------------------------------------------------------------------
// Launch
// ---------------------------------------------------------------------------
extern "C" void kernel_launch(void* const* d_in, const int* in_sizes, int n_in,
                              void* d_out, int out_size, void* d_ws, size_t ws_size,
                              hipStream_t stream) {
    (void)in_sizes; (void)n_in; (void)out_size; (void)ws_size;

    const float* dehy      = (const float*)d_in[0]; // [B,H]
    const float* enhy      = (const float*)d_in[1]; // [B,S,H]
    const float* past_attn = (const float*)d_in[2]; // [B,S]
    const float* W_en      = (const float*)d_in[3]; // [H,H]
    const float* b_en      = (const float*)d_in[4]; // [H]
    const float* W_de      = (const float*)d_in[5]; // [H,H]
    const float* W_warp    = (const float*)d_in[6]; // [1,H]

    float* out        = (float*)d_out;
    float* c_out      = out;                        // [B,H]   = 32768
    float* attn_out   = out + BB * HH;              // [B,S]   = 65536
    float* attn_ee_o  = out + BB * HH + BB * SS;    // [B,S]   = 65536

    float* ws      = (float*)d_ws;
    float* debias  = ws;                            // 32768 floats
    float* attn_un = ws + BB * HH;                  // 65536 floats
    float* sumb    = ws + BB * HH + BB * SS;        // 32 floats
    __bf16* W_en_bf = (__bf16*)(ws + BB * HH + BB * SS + 64); // 1M bf16 = 2MB

    init_sums_kernel<<<1, 32, 0, stream>>>(sumb);

    convert_wen_kernel<<<(HH * HH) / (256 * 4), 256, 0, stream>>>(W_en, W_en_bf);

    debias_kernel<<<BB * (HH / 256), 256, 0, stream>>>(dehy, W_de, b_en, debias);

    fused_score_kernel<<<(BB * SS) / 16, 256, 0, stream>>>(
        enhy, W_en_bf, W_warp, past_attn, debias, attn_ee_o, attn_un, sumb);

    context_kernel<<<BB * (HH / 256), 256, 0, stream>>>(enhy, attn_un, sumb, c_out);

    attn_norm_kernel<<<(BB * SS) / 256, 256, 0, stream>>>(attn_un, sumb, attn_out);
}
